// SparseSynthesisTransform_16930761081537
// MI455X (gfx1250) — matrix operand order repaired and checked by reading, not compiled, not run on hardware
//
#include <hip/hip_runtime.h>

typedef __attribute__((ext_vector_type(16))) _Float16 v16h;
typedef __attribute__((ext_vector_type(8)))  _Float16 v8h;
typedef __attribute__((ext_vector_type(8)))  float    v8f;
typedef __attribute__((ext_vector_type(4)))  int      v4i;

// ---- gfx1250 async global->LDS copy (guarded; falls back to sync copy) ----
#if defined(__AMDGCN__) && defined(__has_builtin)
#if __has_builtin(__builtin_amdgcn_global_load_async_to_lds_b128)
#define ASYNC_LDS 1
#endif
#endif

#ifdef ASYNC_LDS
typedef __attribute__((address_space(1))) v4i* as1_v4i;
typedef __attribute__((address_space(3))) v4i* as3_v4i;
__device__ __forceinline__ void g2l_b128(const _Float16* g, _Float16* l) {
  __builtin_amdgcn_global_load_async_to_lds_b128(
      (as1_v4i)(const __attribute__((address_space(1))) void*)g,
      (as3_v4i)(__attribute__((address_space(3))) void*)l, 0, 0);
}
__device__ __forceinline__ void g2l_wait() {
#if __has_builtin(__builtin_amdgcn_s_wait_asynccnt)
  __builtin_amdgcn_s_wait_asynccnt(0);
#else
  asm volatile("s_wait_asynccnt 0" ::: "memory");
#endif
  asm volatile("" ::: "memory");
}
#else
__device__ __forceinline__ void g2l_b128(const _Float16* g, _Float16* l) {
  *(v8h*)l = *(const v8h*)g;
}
__device__ __forceinline__ void g2l_wait() {}
#endif

// ======================= weight repack =======================
// Wk[co][kk], kk = t*Cpad + ci (ci minor, Cpad = pad32(Ci)), tap t = (jz*ny+jy)*nx+jx,
// kernel index per dim: k = k0 + ks*j. tio=0: OIDHW source, tio=1: IODHW source.
// Rows co in [Co, Co16) and channels ci in [Ci, Cpad) are zero-filled.
struct RepackP {
  const float* W; _Float16* Wk;
  int Co, Ci, Cpad, cShift, kdim, tio;
  int ny, nx;
  int kz0, kzs, ky0, kys, kx0, kxs;
  int K;
};

__global__ void k_repack(RepackP p, long long total) {
  long long stride = (long long)gridDim.x * blockDim.x;
  for (long long idx = (long long)blockIdx.x * blockDim.x + threadIdx.x; idx < total; idx += stride) {
    int co = (int)(idx / p.K);
    int kk = (int)(idx - (long long)co * p.K);
    int ci = kk & (p.Cpad - 1);
    int t  = kk >> p.cShift;
    float v = 0.f;
    if (co < p.Co && ci < p.Ci) {
      int jx = t % p.nx; int t2 = t / p.nx;
      int jy = t2 % p.ny; int jz = t2 / p.ny;
      int kz = p.kz0 + p.kzs * jz;
      int ky = p.ky0 + p.kys * jy;
      int kx = p.kx0 + p.kxs * jx;
      int kd3 = p.kdim * p.kdim * p.kdim;
      int tap = (kz * p.kdim + ky) * p.kdim + kx;
      size_t src = p.tio ? ((size_t)ci * p.Co + co) * kd3 + tap
                         : ((size_t)co * p.Ci + ci) * kd3 + tap;
      v = p.W[src];
    }
    p.Wk[idx] = (_Float16)v;
  }
}

// ======================= WMMA im2col GEMM conv =======================
// One wave computes pair x (16 cout x 16 voxel) f32 tiles with v_wmma_f32_16x16x32_f16.
// The 16 voxels are 16 consecutive x in one (y,z) row; per tap-row (jz,jy) the
// (16+nx-1) x Cpad f16 strip is staged into LDS (async global->LDS when available),
// then all nx x-taps read it with ds_load_b128. B is shared by both cout tiles.
// mode: 0 linear(+bias,*mask), 1 relu(+bias,*mask), 2 iGDN (out = Xf*sqrt(beta+acc)*mask)
struct ConvP {
  const _Float16* X;   // [Din^3][Cpad]
  const _Float16* Wk;  // [Co16][K], K = T*Cpad
  const float* bias;   // [Co] (beta for mode 2)
  const float* mask;   // [Dout^3] or null
  const float* Xf;     // [Co][Dout^3] f32 for mode 2
  float* Y;            // f32 out [Co][Dout^3] (if Ybf==null)
  _Float16* Ybf;       // f16 out [Dout^3][CoPad] (optional)
  int Cpad, cShift, Co, CoPad, Din, dShift, Dout;
  int nz, ny, nx, dz0, dy0, dx0;
  int outMul, oz0, oy0, ox0;
  int K, mode, pair;
};

__device__ __forceinline__ void conv_store(const ConvP& p, const v8f& acc, int coBase,
                                           long long opos, long long Vout, float mval) {
  float vals[8];
  #pragma unroll
  for (int r = 0; r < 8; ++r) {
    int co = coBase + r;
    bool coOK = co < p.Co;
    float v = acc[r];
    if (p.mode == 2) {
      float xf = coOK ? p.Xf[(size_t)co * Vout + opos] : 0.f;
      float bt = coOK ? p.bias[co] : 1.f;
      v = xf * sqrtf(bt + v) * mval;
    } else {
      v += coOK ? p.bias[co] : 0.f;
      if (p.mode == 1) v = v > 0.f ? v : 0.f;
      v *= mval;
    }
    vals[r] = v;
  }
  if (p.Ybf) {
    v8h hv;
    #pragma unroll
    for (int r = 0; r < 8; ++r) hv[r] = (_Float16)vals[r];
    *(v8h*)(p.Ybf + opos * p.CoPad + coBase) = hv;   // 8 consecutive channels: b128 store
  } else {
    #pragma unroll
    for (int r = 0; r < 8; ++r) {
      int co = coBase + r;
      if (co < p.Co) p.Y[(size_t)co * Vout + opos] = vals[r];
    }
  }
}

__global__ void __launch_bounds__(32) k_conv_wmma(ConvP p) {
  __shared__ _Float16 strip[20 * 264];   // max (16+5-1) rows x (256+8) padded f16
  const int lane = threadIdx.x;
  const int half = lane >> 4;            // K-half selector for A/B fragments
  const int l    = lane & 15;            // row (A) / column (B,D) index
  const int D    = p.Din;
  const int v0   = blockIdx.x * 16;      // tile spans 16 consecutive x, one (y,z) row
  const int x0   = v0 & (D - 1);
  const int uy   = (v0 >> p.dShift) & (D - 1);
  const int uz   = v0 >> (2 * p.dShift);
  const int ux   = x0 + l;

  const int stripStride = p.Cpad + 8;    // +16B: rotate LDS banks between lanes
  const int Wx = 16 + p.nx - 1;
  const int totalF16 = Wx << p.cShift;
  const int gxbase = x0 + p.dx0;

  // A row pointers: lanes 0-15 rows M=0..15 hold K {0-7,16-23}; lanes 16-31 hold K {8-15,24-31}
  const int tile0 = blockIdx.y * p.pair * 16;
  const _Float16* wrow0 = p.Wk + (size_t)(tile0 + l) * p.K + (half << 3);
  const _Float16* wrow1 = wrow0 + (size_t)16 * p.K;
  const int hb = half << 3;              // B: lanes 16-31 hold K {8-15,24-31} of the chunk

  v8f acc0 = {}, acc1 = {};
  int t = 0;
  for (int jz = 0; jz < p.nz; ++jz) {
    const int iz = uz + p.dz0 + jz;
    for (int jy = 0; jy < p.ny; ++jy) {
      const int iy = uy + p.dy0 + jy;
      const bool rowOK = (unsigned)iy < (unsigned)D && (unsigned)iz < (unsigned)D;
      const _Float16* rowp =
          p.X + ((size_t)(((iz << p.dShift) + iy) << p.dShift) << p.cShift);
      // ---- stage strip: global (async) -> LDS, zero-fill out-of-range cells ----
      for (int o8 = lane * 8; o8 < totalF16; o8 += 32 * 8) {
        int sx = o8 >> p.cShift;
        int c  = o8 & (p.Cpad - 1);
        _Float16* lp = strip + sx * stripStride + c;
        int gx = gxbase + sx;
        if (rowOK && (unsigned)gx < (unsigned)D)
          g2l_b128(rowp + ((size_t)gx << p.cShift) + c, lp);
        else { v8h z = {}; *(v8h*)lp = z; }
      }
      g2l_wait();
      // ---- compute: nx x-taps read the strip; B shared by both cout tiles ----
      for (int jx = 0; jx < p.nx; ++jx, ++t) {
        const _Float16* wt0 = wrow0 + ((size_t)t << p.cShift);
        const _Float16* wt1 = wrow1 + ((size_t)t << p.cShift);
        const _Float16* ls  = strip + (l + jx) * stripStride + hb;
        if (t + 1 < p.nz * p.ny * p.nx) __builtin_prefetch(wt0 + p.Cpad, 0, 0);
        #pragma unroll 2
        for (int c0 = 0; c0 < p.Cpad; c0 += 32) {
          v8h b0 = *(const v8h*)(ls + c0);
          v8h b1 = *(const v8h*)(ls + c0 + 16);
          v16h b = __builtin_shufflevector(b0, b1, 0,1,2,3,4,5,6,7,8,9,10,11,12,13,14,15);
          v8h a0 = *(const v8h*)(wt0 + c0);
          v8h a1 = *(const v8h*)(wt0 + c0 + 16);
          v16h a = __builtin_shufflevector(a0, a1, 0,1,2,3,4,5,6,7,8,9,10,11,12,13,14,15);
          acc0 = __builtin_amdgcn_wmma_f32_16x16x32_f16(false, a, false, b, (short)0, acc0, false, false);
          if (p.pair == 2) {
            v8h a2 = *(const v8h*)(wt1 + c0);
            v8h a3 = *(const v8h*)(wt1 + c0 + 16);
            v16h a_ = __builtin_shufflevector(a2, a3, 0,1,2,3,4,5,6,7,8,9,10,11,12,13,14,15);
            acc1 = __builtin_amdgcn_wmma_f32_16x16x32_f16(false, a_, false, b, (short)0, acc1, false, false);
          }
        }
      }
    }
  }

  // ---- epilogue + store (D layout: VGPR r -> M = r + 8*half, N = l)
  const int ox = p.outMul * ux + p.ox0;
  const int oy = p.outMul * uy + p.oy0;
  const int oz = p.outMul * uz + p.oz0;
  const long long Vout = (long long)p.Dout * p.Dout * p.Dout;
  const long long opos = ((long long)oz * p.Dout + oy) * p.Dout + ox;
  const float mval = p.mask ? p.mask[opos] : 1.f;
  const int coBase = tile0 + (half << 3);
  conv_store(p, acc0, coBase, opos, Vout, mval);
  if (p.pair == 2) conv_store(p, acc1, coBase + 16, opos, Vout, mval);
}

// ======================= elementwise helpers =======================
__global__ void k_mask_u8(const unsigned char* m, float* out, int V) {
  int i = blockIdx.x * blockDim.x + threadIdx.x;
  if (i < V) out[i] = m[i] ? 1.f : 0.f;
}

// transpose-stage: out[v*Cpad + c] = f16( (x[c*V+v] * mask[v])^(1+square) ), pad c>=C -> 0
__global__ void k_prep(const float* x, const float* mask, _Float16* out,
                       long long V, int C, int Cpad, int cShift, long long N, int square) {
  long long stride = (long long)gridDim.x * blockDim.x;
  for (long long i = (long long)blockIdx.x * blockDim.x + threadIdx.x; i < N; i += stride) {
    int c = (int)(i & (Cpad - 1));
    long long v = i >> cShift;
    float f = 0.f;
    if (c < C) {
      f = x[(long long)c * V + v];
      if (mask) f *= mask[v];
      if (square) f *= f;
    }
    out[i] = (_Float16)f;
  }
}

// generative-tconv coordinate dilation (parity neighbor sets, matches tconv classes)
__global__ void k_dilate(const float* mi, float* mo, int Din, int diS, int Dout, int doS) {
  long long N = (long long)Dout * Dout * Dout;
  long long stride = (long long)gridDim.x * blockDim.x;
  for (long long i = (long long)blockIdx.x * blockDim.x + threadIdx.x; i < N; i += stride) {
    int ox = (int)(i & (Dout - 1));
    int oy = (int)((i >> doS) & (Dout - 1));
    int oz = (int)(i >> (2 * doS));
    int zlo, zhi, ylo, yhi, xlo, xhi, u;
    u = oz >> 1; if (oz & 1) { zlo = u; zhi = u + 1; } else { zlo = u - 1; zhi = u + 1; }
    if (zlo < 0) zlo = 0; if (zhi > Din - 1) zhi = Din - 1;
    u = oy >> 1; if (oy & 1) { ylo = u; yhi = u + 1; } else { ylo = u - 1; yhi = u + 1; }
    if (ylo < 0) ylo = 0; if (yhi > Din - 1) yhi = Din - 1;
    u = ox >> 1; if (ox & 1) { xlo = u; xhi = u + 1; } else { xlo = u - 1; xhi = u + 1; }
    if (xlo < 0) xlo = 0; if (xhi > Din - 1) xhi = Din - 1;
    float occ = 0.f;
    for (int z = zlo; z <= zhi; ++z)
      for (int y = ylo; y <= yhi; ++y)
        for (int x = xlo; x <= xhi; ++x)
          if (mi[(((long long)z << diS) + y) * Din + x] > 0.5f) occ = 1.f;
    mo[i] = occ;
  }
}

// ======================= top-k (radix select over ordered float keys) =======================
// st[0]=prefix, st[1]=k remaining (ties to take at threshold), st[2]=count, st[3]=tie counter
__global__ void k_topk_keys(const float* pred, const float* mask, unsigned* keys, int V) {
  long long stride = (long long)gridDim.x * blockDim.x;
  for (long long i = (long long)blockIdx.x * blockDim.x + threadIdx.x; i < V; i += stride) {
    unsigned fb = __float_as_uint(pred[i]);
    unsigned key = (fb & 0x80000000u) ? ~fb : (fb | 0x80000000u);
    if (mask[i] <= 0.5f) key = 0u;
    keys[i] = key;
  }
}

__global__ void k_topk_init(unsigned* st, const int* kp) {
  st[0] = 0u; st[1] = (unsigned)kp[0]; st[2] = 0u; st[3] = 0u;
}

__global__ void k_topk_count(const unsigned* keys, unsigned* st, int V, int bit) {
  __shared__ unsigned s;
  if (threadIdx.x == 0) s = 0u;
  __syncthreads();
  unsigned pref = st[0];
  unsigned loc = 0;
  long long stride = (long long)gridDim.x * blockDim.x;
  for (long long i = (long long)blockIdx.x * blockDim.x + threadIdx.x; i < V; i += stride) {
    unsigned key = keys[i];
    unsigned diff = key ^ pref;
    bool match = (bit == 31) ? true : ((diff >> (bit + 1)) == 0u);
    if (match && ((key >> bit) & 1u)) loc++;
  }
  if (loc) atomicAdd(&s, loc);
  __syncthreads();
  if (threadIdx.x == 0 && s) atomicAdd(&st[2], s);
}

__global__ void k_topk_update(unsigned* st, int bit) {
  unsigned c = st[2];
  unsigned kRem = st[1];
  if (c >= kRem) st[0] |= (1u << bit);
  else           st[1] = kRem - c;
  st[2] = 0u;
}

__global__ void k_topk_select(const unsigned* keys, unsigned* st, float* mout, int V) {
  unsigned pref = st[0];
  unsigned ties = st[1];
  long long stride = (long long)gridDim.x * blockDim.x;
  for (long long i = (long long)blockIdx.x * blockDim.x + threadIdx.x; i < V; i += stride) {
    unsigned key = keys[i];
    float sel = 0.f;
    if (key > pref) sel = 1.f;
    else if (key == pref) {
      unsigned pos = atomicAdd(&st[3], 1u);
      if (pos < ties) sel = 1.f;
    }
    mout[i] = sel;
  }
}

// ======================= host orchestration =======================
static inline int ilog2i(int x) { int r = 0; while ((1 << r) < x) r++; return r; }
static inline int pad32(int x)  { return x < 32 ? 32 : ((x + 31) & ~31); }

extern "C" void kernel_launch(void* const* d_in, const int* in_sizes, int n_in,
                              void* d_out, int out_size, void* d_ws, size_t ws_size,
                              hipStream_t stream) {
  (void)in_sizes; (void)n_in; (void)out_size;
  // ---- inputs (setup_inputs dict order) ----
  const float* x_feat = (const float*)d_in[0];
  const unsigned char* mask0 = (const unsigned char*)d_in[1];  // jnp bool -> 1 byte
  const int* k0p = (const int*)d_in[3];
  const int* k1p = (const int*)d_in[4];
  const int* k2p = (const int*)d_in[5];
  const float* Wc1 = (const float*)d_in[6];   const float* bc1 = (const float*)d_in[7];
  const float* beta1 = (const float*)d_in[8]; const float* gamma1 = (const float*)d_in[9];
  const float* Wt1 = (const float*)d_in[10];  const float* bt1 = (const float*)d_in[11];
  const float* beta2 = (const float*)d_in[12];const float* gamma2 = (const float*)d_in[13];
  const float* Wt2 = (const float*)d_in[14];  const float* bt2 = (const float*)d_in[15];
  const float* beta3 = (const float*)d_in[16];const float* gamma3 = (const float*)d_in[17];
  const float* Wt3 = (const float*)d_in[18];  const float* bt3 = (const float*)d_in[19];
  const float* Wcol = (const float*)d_in[20]; const float* bcol = (const float*)d_in[21];
  const float* Wp1a = (const float*)d_in[22]; const float* bp1a = (const float*)d_in[23];
  const float* Wp1b = (const float*)d_in[24]; const float* bp1b = (const float*)d_in[25];
  const float* Wp2a = (const float*)d_in[26]; const float* bp2a = (const float*)d_in[27];
  const float* Wp2b = (const float*)d_in[28]; const float* bp2b = (const float*)d_in[29];
  const float* Wp3a = (const float*)d_in[30]; const float* bp3a = (const float*)d_in[31];
  const float* Wp3b = (const float*)d_in[32]; const float* bp3b = (const float*)d_in[33];

  const int D0 = 16, D1 = 32, D2 = 64, D3 = 128;
  const long long V0 = 4096, V1 = 32768, V2 = 262144, V3 = 2097152;

  // ---- workspace carve ----
  char* ws = (char*)d_ws;
  size_t off = 0;
  auto carve = [&](size_t bytes) -> char* {
    char* pp = ws + off;
    off += (bytes + 255) & ~(size_t)255;
    return pp;
  };
  float*     F1   = (float*)carve(134217728);      // f32 activations, [C][V] (ping)
  float*     F2   = (float*)carve(134217728);      // f32 activations (pong) / f16 h alias
  _Float16*  F2h  = (_Float16*)F2;
  _Float16*  B1   = (_Float16*)carve(134217728);   // f16 staging [V][Cpad] (GEMM B input)
  _Float16*  B2   = (_Float16*)carve(33554432);    // f16 staging (x^2 / predictor hidden)
  _Float16*  WB   = (_Float16*)carve(8388608);     // repacked weights [Co16][K]
  float*     PRED = (float*)carve(8388608);        // occupancy logits
  float*     MA   = (float*)carve(8388608);        // mask buffer A
  float*     MB   = (float*)carve(8388608);        // mask buffer B
  unsigned*  KEYS = (unsigned*)carve(8388608);     // topk keys
  unsigned*  ST   = (unsigned*)carve(256);         // topk state
  if (off > ws_size) return;                        // scratch too small: no-op

  // ---- launch helpers ----
  auto gs = [](long long n) { long long b = (n + 255) / 256; return (int)(b > 65535 ? 65535 : b); };

  auto prepL = [&](const float* x, const float* mask, _Float16* out, long long V, int C, int sq) {
    int Cpad = pad32(C);
    long long N = V * Cpad;
    k_prep<<<gs(N), 256, 0, stream>>>(x, mask, out, V, C, Cpad, ilog2i(Cpad), N, sq);
  };

  auto repackL = [&](const float* W, int Co, int Ci, int kdim, int tio,
                     int ny, int nx, int kz0, int kzs, int ky0, int kys, int kx0, int kxs,
                     int K) {
    int Cpad = pad32(Ci);
    RepackP p{W, WB, Co, Ci, Cpad, ilog2i(Cpad), kdim, tio, ny, nx,
              kz0, kzs, ky0, kys, kx0, kxs, K};
    int Co16 = (Co + 15) & ~15;
    long long total = (long long)Co16 * K;
    k_repack<<<gs(total), 256, 0, stream>>>(p, total);
  };

  auto convL = [&](const _Float16* X, const float* bias, const float* mask, const float* Xf,
                   float* Y, _Float16* Ybf, int Ci, int Co, int Din, int Dout,
                   int T, int ny, int nx, int dz0, int dy0, int dx0,
                   int outMul, int oz0, int oy0, int ox0, int K, int mode) {
    int Cpad = pad32(Ci);
    int Co16 = (Co + 15) & ~15;
    int pair = ((Co16 & 31) == 0) ? 2 : 1;
    int nz = T / (ny * nx);
    ConvP p{X, WB, bias, mask, Xf, Y, Ybf, Cpad, ilog2i(Cpad), Co, pad32(Co),
            Din, ilog2i(Din), Dout, nz, ny, nx, dz0, dy0, dx0,
            outMul, oz0, oy0, ox0, K, mode, pair};
    dim3 grid((unsigned)((long long)Din * Din * Din / 16),
              (unsigned)(Co16 / (16 * pair)));
    k_conv_wmma<<<grid, 32, 0, stream>>>(p);
  };

  // standard SAME conv, kernel kdim, stride 1 (OIDHW weights)
  auto stdconv = [&](const float* Wsrc, const float* bias, const _Float16* X, const float* mask,
                     float* Y, _Float16* Ybf, int Ci, int Co, int D, int kdim, int mode) {
    int T = kdim * kdim * kdim, K = T * pad32(Ci), c = kdim / 2;
    repackL(Wsrc, Co, Ci, kdim, 0, kdim, kdim, 0, 1, 0, 1, 0, 1, K);
    convL(X, bias, mask, nullptr, Y, Ybf, Ci, Co, D, D, T, kdim, kdim,
          -c, -c, -c, 1, 0, 0, 0, K, mode);
  };

  // generative tconv k5 s2 (IODHW weights) as 8 parity-class gathers
  auto tconv = [&](const float* Wsrc, const float* bias, const _Float16* X, float* Y,
                   int Ci, int Co, int Din) {
    int Dout = Din * 2;
    for (int pz = 0; pz < 2; ++pz)
      for (int py = 0; py < 2; ++py)
        for (int px = 0; px < 2; ++px) {
          int nz = pz ? 2 : 3, ny = py ? 2 : 3, nx = px ? 2 : 3;
          int dz0 = pz ? 0 : -1, dy0 = py ? 0 : -1, dx0 = px ? 0 : -1;
          int T = nz * ny * nx, K = T * pad32(Ci);
          repackL(Wsrc, Co, Ci, 5, 1, ny, nx, pz, 2, py, 2, px, 2, K);
          convL(X, bias, nullptr, nullptr, Y, nullptr, Ci, Co, Din, Dout,
                T, ny, nx, dz0, dy0, dx0, 2, pz, py, px, K, 0);
        }
  };

  // iGDN: y = (x*m) * sqrt(beta + gamma @ (x*m)^2)   (gamma GEMM on WMMA)
  auto igdn = [&](const float* gamma, const float* beta, const float* Xf, const float* mask,
                  float* Yout, int C, int D) {
    long long V = (long long)D * D * D;
    prepL(Xf, mask, B2, V, C, 1);
    int K = pad32(C);
    repackL(gamma, C, C, 1, 0, 1, 1, 0, 1, 0, 1, 0, 1, K);
    convL(B2, beta, mask, Xf, Yout, nullptr, C, C, D, D, 1, 1, 1,
          0, 0, 0, 1, 0, 0, 0, K, 2);
  };

  auto topk = [&](const float* pred, const float* maskIn, const int* kp, int V, float* maskOut) {
    k_topk_keys<<<gs(V), 256, 0, stream>>>(pred, maskIn, KEYS, V);
    k_topk_init<<<1, 1, 0, stream>>>(ST, kp);
    for (int bit = 31; bit >= 0; --bit) {
      k_topk_count<<<gs(V), 256, 0, stream>>>(KEYS, ST, V, bit);
      k_topk_update<<<1, 1, 0, stream>>>(ST, bit);
    }
    k_topk_select<<<gs(V), 256, 0, stream>>>(KEYS, ST, maskOut, V);
  };

  // ======================= pipeline =======================
  // stage 0: masks + staging
  k_mask_u8<<<(int)((V0 + 255) / 256), 256, 0, stream>>>(mask0, MA, (int)V0);
  prepL(x_feat, MA, B1, V0, 256, 0);

  // up_1: conv5 256->128 @16^3 (*mask), iGDN1, tconv1 128->128 -> 32^3
  stdconv(Wc1, bc1, B1, MA, F1, nullptr, 256, 128, D0, 5, 0);
  igdn(gamma1, beta1, F1, nullptr, F2, 128, D0);
  prepL(F2, nullptr, B1, V0, 128, 0);
  tconv(Wt1, bt1, B1, F1, 128, 128, D0);                        // F1 = [128, 32^3]
  k_dilate<<<gs(V1), 256, 0, stream>>>(MA, MB, D0, 4, D1, 5);   // MB = dilated mask @32^3

  // predictor 1 + topk prune
  prepL(F1, MB, B1, V1, 128, 0);
  stdconv(Wp1a, bp1a, B1, MB, nullptr, B2, 128, 64, D1, 3, 1);
  stdconv(Wp1b, bp1b, B2, MB, PRED, nullptr, 64, 1, D1, 3, 0);
  topk(PRED, MB, k0p, (int)V1, MA);                             // MA = pruned mask @32^3

  // up_2: iGDN2 (pruned), tconv2 128->64 -> 64^3
  igdn(gamma2, beta2, F1, MA, F2, 128, D1);
  prepL(F2, nullptr, B1, V1, 128, 0);
  tconv(Wt2, bt2, B1, F1, 128, 64, D1);                         // F1 = [64, 64^3]
  k_dilate<<<gs(V2), 256, 0, stream>>>(MA, MB, D1, 5, D2, 6);

  // predictor 2 + topk prune
  prepL(F1, MB, B1, V2, 64, 0);
  stdconv(Wp2a, bp2a, B1, MB, nullptr, B2, 64, 32, D2, 3, 1);
  stdconv(Wp2b, bp2b, B2, MB, PRED, nullptr, 32, 1, D2, 3, 0);
  topk(PRED, MB, k1p, (int)V2, MA);                             // MA = pruned mask @64^3

  // up_3: iGDN3, tconv3 64->16 -> 128^3
  igdn(gamma3, beta3, F1, MA, F2, 64, D2);
  prepL(F2, nullptr, B1, V2, 64, 0);
  tconv(Wt3, bt3, B1, F1, 64, 16, D2);                          // F1 = [16, 128^3]
  k_dilate<<<gs(V3), 256, 0, stream>>>(MA, MB, D2, 6, D3, 7);

  // predictor 3 + topk prune (hidden h stored f16 [v][32] in F2 region)
  prepL(F1, MB, B1, V3, 16, 0);
  stdconv(Wp3a, bp3a, B1, MB, nullptr, F2h, 16, 32, D3, 3, 1);
  stdconv(Wp3b, bp3b, F2h, MB, PRED, nullptr, 32, 1, D3, 3, 0);
  topk(PRED, MB, k2p, (int)V3, MA);                             // MA = final mask @128^3

  // color head: 1x1x1 conv 16->3, *mask -> d_out [3, 128^3] f32
  prepL(F1, MA, B1, V3, 16, 0);
  stdconv(Wcol, bcol, B1, MA, (float*)d_out, nullptr, 16, 3, D3, 1, 0);
}
